// AttentionRNN_53266184405106
// MI455X (gfx1250) — matrix operand order repaired and checked by reading, not compiled, not run on hardware
//
#include <hip/hip_runtime.h>
#include <math.h>

typedef _Float16 v16h __attribute__((ext_vector_type(16)));
typedef _Float16 v8h  __attribute__((ext_vector_type(8)));
typedef float    v8f  __attribute__((ext_vector_type(8)));

#define WMMA(a, b, c) \
  __builtin_amdgcn_wmma_f32_16x16x32_f16(false, (a), false, (b), (short)0, (c), false, false)

__device__ __forceinline__ v16h cat8(v8h a, v8h b) {
  return __builtin_shufflevector(a, b, 0,1,2,3,4,5,6,7,8,9,10,11,12,13,14,15);
}

// CDNA5 WMMA f16 A-fragment (16x32): lane m=L&15, hi=L>>4.
// chunk0 = K[kt*32 + hi*16 .. +7], chunk1 = chunk0 + 8  (contiguous K, matches B)
__device__ __forceinline__ v16h load_a_frag(const _Float16* p /* A + m*lda + kt*32 + hi*16 */) {
  v8h c0 = *(const v8h*)p;
  v8h c1 = *(const v8h*)(p + 8);
  return cat8(c0, c1);
}
// B-fragment (32x16): lane n=L&15, hi=L>>4 -> 16 contiguous K at kt*32 + hi*16
__device__ __forceinline__ v16h load_b_frag(const _Float16* p /* B + n*ldb + kt*32 + hi*16 */) {
  return *(const v16h*)p;
}

// monotone float<->uint key for atomicMax-based float max
__device__ __forceinline__ unsigned fkey(float f) {
  unsigned u = __float_as_uint(f);
  return (u & 0x80000000u) ? ~u : (u | 0x80000000u);
}
__device__ __forceinline__ float funkey(unsigned k) {
  return __uint_as_float((k & 0x80000000u) ? (k & 0x7FFFFFFFu) : ~k);
}
__device__ __forceinline__ float sigmoidf_(float x) { return 1.0f / (1.0f + __expf(-x)); }

// ---------------------------------------------------------------- converters
__global__ void f32_to_f16_kernel(const float* __restrict__ src,
                                  _Float16* __restrict__ dst, int n) {
  for (int i = blockIdx.x * blockDim.x + threadIdx.x; i < n; i += gridDim.x * blockDim.x)
    dst[i] = (_Float16)src[i];
}
// dst[c*rows + r] = src[r*cols + c]  (W_note -> [e][d] for contiguous-K B frags)
__global__ void transpose_f16_kernel(const float* __restrict__ src,
                                     _Float16* __restrict__ dst, int rows, int cols) {
  int n = rows * cols;
  for (int i = blockIdx.x * blockDim.x + threadIdx.x; i < n; i += gridDim.x * blockDim.x) {
    int r = i / cols, c = i - r * cols;
    dst[c * rows + r] = (_Float16)src[i];
  }
}
__global__ void zero_f32_kernel(float* p, int n) {
  for (int i = blockIdx.x * blockDim.x + threadIdx.x; i < n; i += gridDim.x * blockDim.x)
    p[i] = 0.0f;
}

// --------------------------------------------- fused gather+conv+tanh+maxpool
// one block per sample (n,b).
// LDS: tokens | fmax | f16 embeddings [274][128]: word w at row w+2;
// rows 0-1 and 258-273 are zero => conv zero-padding + safe garbage l-tiles,
// so all A-fragment LDS loads are unconditional (no exec-mask predication).
__global__ void __launch_bounds__(256)
conv_kernel(const int* __restrict__ toks,      // [64][256][32]
            const float* __restrict__ emb,     // [50000][128]
            const _Float16* __restrict__ wcv,  // [256][384] f16, k-contig
            const float* __restrict__ cb,      // [256]
            _Float16* __restrict__ featH) {    // [2048][256]
  extern __shared__ char smem[];
  int*      ltok = (int*)smem;                  // 1 KB
  unsigned* fmax = (unsigned*)(smem + 1024);    // 1 KB
  _Float16* aE   = (_Float16*)(smem + 2048);    // 70,144 B: [274][128]

  const int sample = blockIdx.x;
  const int note = sample >> 5, bb = sample & 31;
  const int tid = threadIdx.x;

  ltok[tid] = toks[(note * 256 + tid) * 32 + bb];
  fmax[tid] = 0u;  // below fkey of any float
  // zero the 18 padding rows (0,1 and 258..273)
  for (int i = tid; i < 18 * 128; i += 256) {
    int r = i >> 7;
    int row = (r < 2) ? r : (r + 256);
    aE[row * 128 + (i & 127)] = (_Float16)0.0f;
  }
  __syncthreads();

  {  // gather: thread t -> word t (128 dims, float4 vectorized), row t+2
    const float4* src = (const float4*)(emb + (long)ltok[tid] * 128);
    _Float16* dst = aE + (tid + 2) * 128;
#pragma unroll 4
    for (int i = 0; i < 32; ++i) {
      float4 v = src[i];
      dst[i * 4 + 0] = (_Float16)v.x; dst[i * 4 + 1] = (_Float16)v.y;
      dst[i * 4 + 2] = (_Float16)v.z; dst[i * 4 + 3] = (_Float16)v.w;
    }
  }
  __syncthreads();

  const int wave = tid >> 5, lane = tid & 31;
  const int m15 = lane & 15, hi = (lane >> 4) & 1;

  // 17 l-tiles (l = 0..271, valid l < 258) x 16 oc-tiles.
  // Per l-tile: hoist all 12 A-fragments to registers, reuse across oc-tiles;
  // two independent accumulators fill the WMMA hazard slots.
  for (int mt = wave; mt < 17; mt += 8) {
    const int l = mt * 16 + m15;
    v16h af[12];
#pragma unroll
    for (int kt = 0; kt < 12; ++kt) {
      // A[l][k] = lds row (l + k/128), col k%128  (row offset +2 - 2 folded in)
      int k0 = kt * 32 + hi * 16;
      int r0 = l + (k0 >> 7), d0 = k0 & 127;
      int k1 = k0 + 8;
      int r1 = l + (k1 >> 7), d1 = k1 & 127;
      af[kt] = cat8(*(const v8h*)(aE + r0 * 128 + d0),
                    *(const v8h*)(aE + r1 * 128 + d1));
    }
    for (int np = 0; np < 8; ++np) {
      const int nt0 = np * 2;
      const _Float16* bp0 = wcv + (nt0 * 16 + m15) * 384 + hi * 16;
      const _Float16* bp1 = bp0 + 16 * 384;
      v8f acc0 = {}, acc1 = {};
#pragma unroll
      for (int kt = 0; kt < 12; ++kt) {
        v16h b0 = *(const v16h*)(bp0 + kt * 32);
        v16h b1 = *(const v16h*)(bp1 + kt * 32);
        acc0 = WMMA(af[kt], b0, acc0);
        acc1 = WMMA(af[kt], b1, acc1);
      }
      // tanh + max over l (mask garbage l >= 258), for both oc-tiles
#pragma unroll
      for (int q = 0; q < 2; ++q) {
        const v8f& acc = q ? acc1 : acc0;
        const int oc = (nt0 + q) * 16 + m15;
        const float bias = cb[oc];
        float mx = -1e30f;
#pragma unroll
        for (int r = 0; r < 8; ++r) {
          int ll = mt * 16 + hi * 8 + r;
          if (ll < 258) mx = fmaxf(mx, tanhf(acc[r] + bias));
        }
        mx = fmaxf(mx, __shfl_xor(mx, 16));  // partner lane has same oc
        if (hi == 0) atomicMax(&fmax[oc], fkey(mx));
      }
    }
  }
  __syncthreads();
  featH[sample * 256 + tid] = (_Float16)funkey(fmax[tid]);
}

// ------------------------------------------------------- generic WMMA GEMM
// C[m][n] = sum_k A[m][k] * B[n][k]; one 16x32 tile-pair per wave
// (two n-tiles share one A fragment -> independent WMMA chains)
__global__ void __launch_bounds__(256)
gemm_f16f32_kernel(const _Float16* __restrict__ A, const _Float16* __restrict__ B,
                   float* __restrict__ C, int Mtiles, int Npairs, int Ktiles,
                   int lda, int ldb, int ldc) {
  const int wave = threadIdx.x >> 5, lane = threadIdx.x & 31;
  const int tile = blockIdx.x * 8 + wave;
  if (tile >= Mtiles * Npairs) return;
  const int mt = tile / Npairs, np = tile - mt * Npairs;
  const int m15 = lane & 15, hi = (lane >> 4) & 1;
  const _Float16* ap  = A + (mt * 16 + m15) * lda + hi * 16;
  const _Float16* bp0 = B + (np * 32 + m15) * ldb + hi * 16;
  const _Float16* bp1 = bp0 + 16 * ldb;
  v8f acc0 = {}, acc1 = {};
  for (int kt = 0; kt < Ktiles; ++kt) {
    v16h a  = load_a_frag(ap);
    v16h b0 = load_b_frag(bp0);
    v16h b1 = load_b_frag(bp1);
    acc0 = WMMA(a, b0, acc0);
    acc1 = WMMA(a, b1, acc1);
    ap += 32; bp0 += 32; bp1 += 32;
  }
  float* cp = C + (mt * 16 + hi * 8) * ldc + np * 32 + m15;
#pragma unroll
  for (int r = 0; r < 8; ++r) {
    cp[r * ldc]      = acc0[r];
    cp[r * ldc + 16] = acc1[r];
  }
}

// -------------------------------------------------------------- GRU (serial)
// one block per direction; h lives in LDS (f16 A-fragments + f32 copy)
__global__ void __launch_bounds__(512)
gru_kernel(const float* __restrict__ gi,       // [2][64*32][768]
           const _Float16* __restrict__ whhH,  // [2][768][256]
           const float* __restrict__ hidden,   // [2][32][256]
           const float* __restrict__ bihF, const float* __restrict__ bhhF,
           const float* __restrict__ bihB, const float* __restrict__ bhhB,
           float* __restrict__ outNote,        // [64][32][512]
           _Float16* __restrict__ outNoteH) {  // [64][32][512]
  extern __shared__ char smem[];
  _Float16* hA = (_Float16*)smem;               // [32][256] f16  (16 KB)
  float*    hF = (float*)(smem + 16384);        // [32][256] f32  (32 KB)
  float*    gh = (float*)(smem + 49152);        // [32][768] f32  (96 KB)

  const int dir = blockIdx.x;
  const int tid = threadIdx.x;
  const int wave = tid >> 5, lane = tid & 31;
  const int m15 = lane & 15, hi = (lane >> 4) & 1;
  const float* giD = gi + (long)dir * 2048 * 768;
  const _Float16* W = whhH + dir * 768 * 256;
  const float* bi = dir ? bihB : bihF;
  const float* bh = dir ? bhhB : bhhF;
  const float* h0 = hidden + dir * 32 * 256;

  for (int i = tid; i < 32 * 256; i += 512) {
    float v = h0[i];
    hF[i] = v; hA[i] = (_Float16)v;
  }
  __syncthreads();

  for (int s = 0; s < 64; ++s) {
    const int n = dir ? (63 - s) : s;
    // gh = hA @ whh^T : 2 m-tiles x 24 n-tile-pairs, K=256 (8 k-tiles), 16 waves
    for (int p = wave; p < 48; p += 16) {
      const int mt = p / 24, nb = (p - mt * 24) * 2;
      const _Float16* ap  = hA + (mt * 16 + m15) * 256 + hi * 16;
      const _Float16* bp0 = W + (nb * 16 + m15) * 256 + hi * 16;
      const _Float16* bp1 = bp0 + 16 * 256;
      v8f acc0 = {}, acc1 = {};
#pragma unroll
      for (int kt = 0; kt < 8; ++kt) {
        v16h a  = load_a_frag(ap);
        v16h b0 = load_b_frag(bp0);
        v16h b1 = load_b_frag(bp1);
        acc0 = WMMA(a, b0, acc0);
        acc1 = WMMA(a, b1, acc1);
        ap += 32; bp0 += 32; bp1 += 32;
      }
      const int nn = nb * 16 + m15, mb = mt * 16 + hi * 8;
#pragma unroll
      for (int r = 0; r < 8; ++r) {
        gh[(mb + r) * 768 + nn]      = acc0[r];
        gh[(mb + r) * 768 + nn + 16] = acc1[r];
      }
    }
    __syncthreads();
    // gates
    const float* gin = giD + (long)n * 32 * 768;
    for (int i = tid; i < 32 * 256; i += 512) {
      const int b = i >> 8, j = i & 255;
      float gir = gin[b * 768 + j]       + bi[j];
      float giz = gin[b * 768 + 256 + j] + bi[256 + j];
      float gnn = gin[b * 768 + 512 + j] + bi[512 + j];
      float ghr = gh[b * 768 + j]        + bh[j];
      float ghz = gh[b * 768 + 256 + j]  + bh[256 + j];
      float ghn = gh[b * 768 + 512 + j]  + bh[512 + j];
      float r = sigmoidf_(gir + ghr);
      float z = sigmoidf_(giz + ghz);
      float ng = tanhf(gnn + r * ghn);
      float hnew = (1.0f - z) * ng + z * hF[i];
      hF[i] = hnew;
      hA[i] = (_Float16)hnew;
      const long o = (long)(n * 32 + b) * 512 + dir * 256 + j;
      outNote[o]  = hnew;
      outNoteH[o] = (_Float16)hnew;
    }
    __syncthreads();
  }
}

// ------------------------------------------ attention GEMM (tanh+proj fused)
// two n-tiles per wave sharing the A fragment
__global__ void __launch_bounds__(256)
attn_gemm_kernel(const _Float16* __restrict__ A,   // outNoteH [2048][512]
                 const _Float16* __restrict__ Bt,  // W_note^T f16 [512][512]
                 const float* __restrict__ bias, const float* __restrict__ proj,
                 float* __restrict__ attn) {       // [2048]
  const int wave = threadIdx.x >> 5, lane = threadIdx.x & 31;
  const int tile = blockIdx.x * 8 + wave;
  if (tile >= 128 * 16) return;                 // 128 m-tiles x 16 n-pairs
  const int mt = tile >> 4, np = tile & 15;
  const int m15 = lane & 15, hi = (lane >> 4) & 1;
  const _Float16* ap  = A + (mt * 16 + m15) * 512 + hi * 16;
  const _Float16* bp0 = Bt + (np * 32 + m15) * 512 + hi * 16;
  const _Float16* bp1 = bp0 + 16 * 512;
  v8f acc0 = {}, acc1 = {};
#pragma unroll
  for (int kt = 0; kt < 16; ++kt) {
    v16h a  = load_a_frag(ap);
    v16h b0 = load_b_frag(bp0);
    v16h b1 = load_b_frag(bp1);
    acc0 = WMMA(a, b0, acc0);
    acc1 = WMMA(a, b1, acc1);
    ap += 32; bp0 += 32; bp1 += 32;
  }
#pragma unroll
  for (int q = 0; q < 2; ++q) {
    const v8f& acc = q ? acc1 : acc0;
    const int e = (np * 2 + q) * 16 + m15;
    const float pr = proj[e], bs = bias[e];
#pragma unroll
    for (int r = 0; r < 8; ++r) {
      float v = tanhf(acc[r] + bs) * pr;
      for (int off = 1; off < 16; off <<= 1) v += __shfl_xor(v, off);
      if (m15 == 0) atomicAdd(&attn[mt * 16 + hi * 8 + r], v);
    }
  }
}

// ------------------------------------- softmax over notes + pooled head
__global__ void __launch_bounds__(64)
finalize_kernel(const float* __restrict__ attn,     // [2048], m = n*32+b
                const float* __restrict__ outNote,  // [64][32][512]
                const float* __restrict__ linW, const float* __restrict__ linB,
                float* __restrict__ attnNorm,       // [32][64]
                float* __restrict__ vecOut,         // [32][512]
                float* __restrict__ mapOut) {       // [32][2]
  __shared__ float p[64];
  __shared__ float red[64];
  const int b = blockIdx.x, t = threadIdx.x;
  const float a = attn[t * 32 + b];
  red[t] = a; __syncthreads();
  for (int s = 32; s > 0; s >>= 1) { if (t < s) red[t] = fmaxf(red[t], red[t + s]); __syncthreads(); }
  const float mx = red[0]; __syncthreads();
  const float e = __expf(a - mx);
  red[t] = e; __syncthreads();
  for (int s = 32; s > 0; s >>= 1) { if (t < s) red[t] += red[t + s]; __syncthreads(); }
  const float pn = e / red[0];
  p[t] = pn;
  attnNorm[b * 64 + t] = pn;
  __syncthreads();
  for (int d = t; d < 512; d += 64) {
    float acc = 0.0f;
    for (int n = 0; n < 64; ++n) acc += p[n] * outNote[(long)(n * 32 + b) * 512 + d];
    vecOut[b * 512 + d] = acc;
  }
  __syncthreads();
  if (t < 2) {
    float acc = linB[t];
    for (int d = 0; d < 512; ++d) acc += vecOut[b * 512 + d] * linW[t * 512 + d];
    mapOut[b * 2 + t] = acc;
  }
}

// ---------------------------------------------------------------------- host
extern "C" void kernel_launch(void* const* d_in, const int* in_sizes, int n_in,
                              void* d_out, int out_size, void* d_ws, size_t ws_size,
                              hipStream_t stream) {
  (void)in_sizes; (void)n_in; (void)out_size; (void)ws_size;
  const int*   mini_batch = (const int*)d_in[0];
  const float* hidden     = (const float*)d_in[1];
  const float* embed      = (const float*)d_in[2];
  const float* conv_w     = (const float*)d_in[3];
  const float* conv_b     = (const float*)d_in[4];
  const float* w_ih_f     = (const float*)d_in[5];
  const float* w_hh_f     = (const float*)d_in[6];
  const float* b_ih_f     = (const float*)d_in[7];
  const float* b_hh_f     = (const float*)d_in[8];
  const float* w_ih_b     = (const float*)d_in[9];
  const float* w_hh_b     = (const float*)d_in[10];
  const float* b_ih_b     = (const float*)d_in[11];
  const float* b_hh_b     = (const float*)d_in[12];
  const float* W_note     = (const float*)d_in[13];
  const float* bias_note  = (const float*)d_in[14];
  const float* proj_note  = (const float*)d_in[15];
  const float* lin_w      = (const float*)d_in[16];
  const float* lin_b      = (const float*)d_in[17];

  float* out      = (float*)d_out;
  float* outNote  = out;                // 64*32*512
  float* attnNorm = out + 1048576;      // 32*64
  float* vecOut   = attnNorm + 2048;    // 32*512
  float* mapOut   = vecOut + 16384;     // 32*2

  char* ws = (char*)d_ws;
  size_t off = 0;
  auto alloc = [&](size_t bytes) -> void* {
    void* p = ws + off;
    off += (bytes + 255) & ~(size_t)255;
    return p;
  };
  _Float16* wcvH     = (_Float16*)alloc(98304u * 2);       // conv_w f16 [256][384]
  _Float16* wihH     = (_Float16*)alloc(2u * 196608 * 2);  // [dir][768][256]
  _Float16* whhH     = (_Float16*)alloc(2u * 196608 * 2);  // [dir][768][256]
  _Float16* WnT      = (_Float16*)alloc(262144u * 2);      // W_note^T [512][512]
  _Float16* featH    = (_Float16*)alloc(524288u * 2);      // [2048][256]
  float*    gi       = (float*)alloc(2u * 2048 * 768 * 4); // precomputed input gates
  _Float16* outNoteH = (_Float16*)alloc(1048576u * 2);     // [2048][512]
  float*    attn     = (float*)alloc(2048u * 4);

  const int convLds = 2048 + 274 * 128 * 2;          // 72,192 B
  const int gruLds  = 16384 + 32768 + 98304;         // 147,456 B
  hipFuncSetAttribute((const void*)conv_kernel, hipFuncAttributeMaxDynamicSharedMemorySize, convLds);
  hipFuncSetAttribute((const void*)gru_kernel,  hipFuncAttributeMaxDynamicSharedMemorySize, gruLds);

  // weight conversions to f16 (tiny)
  f32_to_f16_kernel<<<96, 256, 0, stream>>>(conv_w, wcvH, 98304);
  f32_to_f16_kernel<<<192, 256, 0, stream>>>(w_ih_f, wihH, 196608);
  f32_to_f16_kernel<<<192, 256, 0, stream>>>(w_ih_b, wihH + 196608, 196608);
  f32_to_f16_kernel<<<192, 256, 0, stream>>>(w_hh_f, whhH, 196608);
  f32_to_f16_kernel<<<192, 256, 0, stream>>>(w_hh_b, whhH + 196608, 196608);
  transpose_f16_kernel<<<256, 256, 0, stream>>>(W_note, WnT, 512, 512);
  zero_f32_kernel<<<8, 256, 0, stream>>>(attn, 2048);

  // fused gather + conv + tanh + maxpool -> featH
  conv_kernel<<<2048, 256, convLds, stream>>>(mini_batch, embed, wcvH, conv_b, featH);

  // gi = feat @ w_ih^T for all steps, both directions (M=2048,N=768,K=256)
  gemm_f16f32_kernel<<<384, 256, 0, stream>>>(featH, wihH, gi,
                                              128, 24, 8, 256, 256, 768);
  gemm_f16f32_kernel<<<384, 256, 0, stream>>>(featH, wihH + 196608, gi + 2048 * 768,
                                              128, 24, 8, 256, 256, 768);

  // sequential bidirectional GRU (one persistent block per direction)
  gru_kernel<<<2, 512, gruLds, stream>>>(gi, whhH, hidden,
                                         b_ih_f, b_hh_f, b_ih_b, b_hh_b,
                                         outNote, outNoteH);

  // attention scores: tanh(out_note @ W_note + b) . proj  (M=2048,N=512,K=512)
  attn_gemm_kernel<<<256, 256, 0, stream>>>(outNoteH, WnT, bias_note, proj_note, attn);

  // softmax over notes, pooled vector, final linear
  finalize_kernel<<<32, 64, 0, stream>>>(attn, outNote, lin_w, lin_b,
                                         attnNorm, vecOut, mapOut);
}
